// GAU_86775519248998
// MI455X (gfx1250) — matrix operand order repaired and checked by reading, not compiled, not run on hardware
//
#include <hip/hip_runtime.h>
#include <hip/hip_bf16.h>
#include <math.h>

#define BATCH 8
#define SEQ   2048
#define DIMX  512
#define QKD   128
#define HID   1024

typedef _Float16 half8_t  __attribute__((ext_vector_type(8)));
typedef _Float16 half16_t __attribute__((ext_vector_type(16)));
typedef float    float8_t __attribute__((ext_vector_type(8)));

__device__ __forceinline__ half16_t frag2(const _Float16* p0, const _Float16* p1) {
    half8_t a = *(const half8_t*)p0;
    half8_t b = *(const half8_t*)p1;
    return __builtin_shufflevector(a, b, 0,1,2,3,4,5,6,7,8,9,10,11,12,13,14,15);
}

__device__ __forceinline__ float8_t wmma_f16(half16_t a, half16_t b, float8_t c) {
    return __builtin_amdgcn_wmma_f32_16x16x32_f16(false, a, false, b, (short)0, c, false, false);
}

__device__ __forceinline__ float silu(float v) { return v / (1.f + expf(-v)); }

// ---------------- prep: weight transpose (K x N f32 -> N x K f16) ----------------
__global__ void transpose_f16_kernel(const float* __restrict__ src, _Float16* __restrict__ dst,
                                     int K, int N) {
    int idx = blockIdx.x * blockDim.x + threadIdx.x;
    if (idx >= K * N) return;
    int n = idx / K, k = idx % K;
    dst[idx] = (_Float16)src[(size_t)k * N + n];
}

// ---------------- prep: T5 relative bias table, index = (j - i) + (SEQ-1) ----------------
__global__ void bias_table_kernel(const float* __restrict__ tbl, float* __restrict__ biasTab) {
    int idx = blockIdx.x * blockDim.x + threadIdx.x;
    if (idx >= 2 * SEQ - 1) return;
    int rel = idx - (SEQ - 1);      // j - i
    int n = -rel;                   // reference: n = -rel_pos
    int ret = 0;
    if (n < 0) { ret = 16; n = -n; }
    int v;
    if (n < 8) v = n;
    else {
        float fv = logf((float)n / 8.f) / logf(128.f / 8.f) * 8.f;
        v = 8 + (int)fv;
        if (v > 15) v = 15;
    }
    biasTab[idx] = tbl[ret + v] * sqrtf((float)QKD);
}

// ---------------- LayerNorm + token-shift + /(qk_s*hidden_s) -> f16 ----------------
__global__ void ln_shift_kernel(const float* __restrict__ x, const float* __restrict__ g,
                                const float* __restrict__ be, const float* __restrict__ qs,
                                const float* __restrict__ hs, _Float16* __restrict__ normed) {
    int row = blockIdx.x;             // b*SEQ + s
    int s = row & (SEQ - 1);
    int t = threadIdx.x;
    int wid = t >> 5, lane = t & 31;
    __shared__ float r_sum[8], r_sq[8];
    __shared__ float stats[4];
    const float* xr = x + (size_t)row * DIMX;
    float a0 = xr[t], a1 = xr[t + 256];
    float p0 = 0.f, p1 = 0.f;
    if (s > 0) { p0 = xr[t - DIMX]; p1 = xr[t + 256 - DIMX]; }

    // current row stats
    float sum = a0 + a1, sq = a0 * a0 + a1 * a1;
    for (int o = 16; o > 0; o >>= 1) { sum += __shfl_down(sum, o); sq += __shfl_down(sq, o); }
    if (lane == 0) { r_sum[wid] = sum; r_sq[wid] = sq; }
    __syncthreads();
    if (t == 0) {
        float ts = 0.f, tq = 0.f;
        for (int i = 0; i < 8; i++) { ts += r_sum[i]; tq += r_sq[i]; }
        float mu = ts / DIMX;
        stats[0] = mu; stats[1] = rsqrtf(tq / DIMX - mu * mu + 1e-5f);
    }
    __syncthreads();
    // previous row stats (zeros when s==0; result unused)
    sum = p0 + p1; sq = p0 * p0 + p1 * p1;
    for (int o = 16; o > 0; o >>= 1) { sum += __shfl_down(sum, o); sq += __shfl_down(sq, o); }
    if (lane == 0) { r_sum[wid] = sum; r_sq[wid] = sq; }
    __syncthreads();
    if (t == 0) {
        float ts = 0.f, tq = 0.f;
        for (int i = 0; i < 8; i++) { ts += r_sum[i]; tq += r_sq[i]; }
        float mu = ts / DIMX;
        stats[2] = mu; stats[3] = rsqrtf(tq / DIMX - mu * mu + 1e-5f);
    }
    __syncthreads();
    float mu_c = stats[0], rs_c = stats[1], mu_p = stats[2], rs_p = stats[3];
    int d0 = t, d1 = t + 256;
    float v0 = (s > 0) ? ((p0 - mu_p) * rs_p * g[d0] + be[d0]) : 0.f;   // shifted half
    float v1 = (a1 - mu_c) * rs_c * g[d1] + be[d1];                     // pass-through half
    v0 /= (qs[d0] * hs[d0]);
    v1 /= (qs[d1] * hs[d1]);
    _Float16* nr = normed + (size_t)row * DIMX;
    nr[d0] = (_Float16)v0;
    nr[d1] = (_Float16)v1;
}

// ---------------- GEMM: normed(BSx512) @ WhT -> silu -> vT(f16, d-major) & gate(f16) ---------
__global__ void gemm_hidden_kernel(const _Float16* __restrict__ normed,
                                   const _Float16* __restrict__ WhT,
                                   const float* __restrict__ bh,
                                   _Float16* __restrict__ vT,
                                   _Float16* __restrict__ gate) {
    int w = threadIdx.x >> 5, l = threadIdx.x & 31;
    int h = l >> 4, nl = l & 15;
    int gw = blockIdx.x * 8 + w;
    int mt = gw >> 5;              // 1024 M tiles
    int n0 = (gw & 31) * 64;       // 32 N-slabs of 64
    const _Float16* Arow = normed + ((size_t)mt * 16 + nl) * DIMX;
    float8_t acc[4] = {};
    for (int kc = 0; kc < 16; kc++) {
        int kb = kc * 32;
        const _Float16* ap = Arow + kb + h * 8;
        half16_t A = frag2(ap, ap + 16);
#pragma unroll
        for (int j = 0; j < 4; j++) {
            int n = n0 + j * 16 + nl;
            const _Float16* bp = WhT + (size_t)n * DIMX + kb + h * 8;
            half16_t B = frag2(bp, bp + 16);
            acc[j] = wmma_f16(A, B, acc[j]);
        }
    }
    int b = (mt * 16) / SEQ;
    int srow = (mt * 16) & (SEQ - 1);
#pragma unroll
    for (int j = 0; j < 4; j++) {
        int n = n0 + j * 16 + nl;
        float bn = bh[n];
        if (n < HID) {
            half8_t hv;
#pragma unroll
            for (int r = 0; r < 8; r++) hv[r] = (_Float16)silu(acc[j][r] + bn);
            *(half8_t*)(vT + ((size_t)b * HID + n) * SEQ + srow + 8 * h) = hv;
        } else {
            int d = n - HID;
#pragma unroll
            for (int r = 0; r < 8; r++) {
                size_t m = (size_t)mt * 16 + r + 8 * h;
                gate[m * HID + d] = (_Float16)silu(acc[j][r] + bn);
            }
        }
    }
}

// ---------------- GEMM: normed @ WqkT -> silu -> qk32 (f32) ----------------
__global__ void gemm_qk_kernel(const _Float16* __restrict__ normed,
                               const _Float16* __restrict__ WqkT,
                               const float* __restrict__ bq, float* __restrict__ qk32) {
    int w = threadIdx.x >> 5, l = threadIdx.x & 31;
    int h = l >> 4, nl = l & 15;
    int gw = blockIdx.x * 8 + w;
    int mt = gw >> 1;
    int n0 = (gw & 1) * 64;
    const _Float16* Arow = normed + ((size_t)mt * 16 + nl) * DIMX;
    float8_t acc[4] = {};
    for (int kc = 0; kc < 16; kc++) {
        int kb = kc * 32;
        const _Float16* ap = Arow + kb + h * 8;
        half16_t A = frag2(ap, ap + 16);
#pragma unroll
        for (int j = 0; j < 4; j++) {
            int n = n0 + j * 16 + nl;
            const _Float16* bp = WqkT + (size_t)n * DIMX + kb + h * 8;
            half16_t B = frag2(bp, bp + 16);
            acc[j] = wmma_f16(A, B, acc[j]);
        }
    }
#pragma unroll
    for (int j = 0; j < 4; j++) {
        int n = n0 + j * 16 + nl;
        float bn = bq[n];
#pragma unroll
        for (int r = 0; r < 8; r++) {
            size_t m = (size_t)mt * 16 + r + 8 * h;
            qk32[m * QKD + n] = silu(acc[j][r] + bn);
        }
    }
}

// ---------------- q/k scale + rotary -> f16 ----------------
__global__ void rotary_kernel(const float* __restrict__ qk32, const float* __restrict__ osg,
                              const float* __restrict__ osb,
                              _Float16* __restrict__ q, _Float16* __restrict__ k) {
    int row = blockIdx.x;
    int d = threadIdx.x;
    int s = row & (SEQ - 1);
    size_t base = (size_t)row * QKD;
    float t = qk32[base + d];
    float qv = t * osg[d] + osb[d];
    float kv = t * osg[QKD + d] + osb[QKD + d];
    float qo = qv, ko = kv;
    if (d < 32) {
        int dp = d ^ 16;
        float tp = qk32[base + dp];
        float qp = tp * osg[dp] + osb[dp];
        float kp = tp * osg[QKD + dp] + osb[QKD + dp];
        int f = d & 15;
        float ang = (float)s * powf(10000.f, -(2.f * (float)f) / 32.f);
        float c = cosf(ang), sn = sinf(ang);
        if (d < 16) { qo = qv * c - qp * sn; ko = kv * c - kp * sn; }
        else        { qo = qv * c + qp * sn; ko = kv * c + kp * sn; }
    }
    q[base + d] = (_Float16)qo;
    k[base + d] = (_Float16)ko;
}

// ---------------- fused attention: sim=q@k^T + bias, relu(sim/S)^2, @v, *gate, /out_s ------
__global__ void attn_kernel(const _Float16* __restrict__ qg, const _Float16* __restrict__ kg,
                            const _Float16* __restrict__ vT, const _Float16* __restrict__ gate,
                            const float* __restrict__ biasTab, const float* __restrict__ out_s,
                            _Float16* __restrict__ out_pre) {
    __shared__ __align__(16) _Float16 probs[16 * 272];   // 16 queries x 256 keys (+pad)
    int w = threadIdx.x >> 5, l = threadIdx.x & 31;
    int h = l >> 4, nl = l & 15;
    int b = blockIdx.x >> 7;
    int i0 = (blockIdx.x & 127) * 16;
    // resident q fragments for this block's 16 queries (K = 128 => 4 chunks)
    const _Float16* qbase = qg + ((size_t)b * SEQ + i0 + nl) * QKD;
    half16_t qf[4];
#pragma unroll
    for (int c = 0; c < 4; c++) {
        const _Float16* p = qbase + c * 32 + h * 8;
        qf[c] = frag2(p, p + 16);
    }
    float8_t acc[8] = {};   // 16 x 128 output columns per wave
    for (int jb = 0; jb < SEQ; jb += 256) {
        // stage 1: each wave computes 2 sim tiles -> probs in LDS
#pragma unroll
        for (int st = 0; st < 2; st++) {
            int jt = 2 * w + st;
            int j0 = jb + jt * 16;
            const _Float16* krow = kg + ((size_t)b * SEQ + j0 + nl) * QKD;
            float8_t sim = {};
#pragma unroll
            for (int c = 0; c < 4; c++) {
                const _Float16* p = krow + c * 32 + h * 8;
                half16_t B = frag2(p, p + 16);
                sim = wmma_f16(qf[c], B, sim);
            }
#pragma unroll
            for (int r = 0; r < 8; r++) {
                int i = i0 + r + 8 * h;
                int j = j0 + nl;
                float v = sim[r] + biasTab[j - i + (SEQ - 1)];
                v = v * (1.f / (float)SEQ);
                v = v > 0.f ? v * v : 0.f;
                probs[(r + 8 * h) * 272 + jt * 16 + nl] = (_Float16)v;
            }
        }
        __syncthreads();
        // stage 2: probs(16x256) @ v(256 x 1024); each wave owns 128 d-columns
        for (int c2 = 0; c2 < 8; c2++) {
            int kb2 = c2 * 32;
            const _Float16* ap = &probs[nl * 272 + kb2 + h * 8];
            half16_t A = frag2(ap, ap + 16);
#pragma unroll
            for (int t = 0; t < 8; t++) {
                int d = w * 128 + t * 16 + nl;
                const _Float16* vp = vT + ((size_t)b * HID + d) * SEQ + jb + kb2 + h * 8;
                half16_t B = frag2(vp, vp + 16);
                acc[t] = wmma_f16(A, B, acc[t]);
            }
        }
        __syncthreads();
    }
    // epilogue: * gate / out_s -> f16 out_pre (row-major, A-matrix of final GEMM)
#pragma unroll
    for (int t = 0; t < 8; t++) {
        int d = w * 128 + t * 16 + nl;
        float osd = out_s[d];
#pragma unroll
        for (int r = 0; r < 8; r++) {
            size_t m = (size_t)b * SEQ + i0 + r + 8 * h;
            float gv = (float)gate[m * HID + d];
            out_pre[m * HID + d] = (_Float16)(acc[t][r] * gv / osd);
        }
    }
}

// ---------------- final GEMM: out_pre(BSx1024) @ WoT + b_out + x -> f32 ----------------
__global__ void gemm_out_kernel(const _Float16* __restrict__ outp,
                                const _Float16* __restrict__ WoT,
                                const float* __restrict__ bo, const float* __restrict__ x,
                                float* __restrict__ out) {
    int w = threadIdx.x >> 5, l = threadIdx.x & 31;
    int h = l >> 4, nl = l & 15;
    int gw = blockIdx.x * 8 + w;
    int mt = gw >> 3;
    int n0 = (gw & 7) * 64;
    const _Float16* Arow = outp + ((size_t)mt * 16 + nl) * HID;
    float8_t acc[4] = {};
    for (int kc = 0; kc < 32; kc++) {
        int kb = kc * 32;
        const _Float16* ap = Arow + kb + h * 8;
        half16_t A = frag2(ap, ap + 16);
#pragma unroll
        for (int j = 0; j < 4; j++) {
            int n = n0 + j * 16 + nl;
            const _Float16* bp = WoT + (size_t)n * HID + kb + h * 8;
            half16_t B = frag2(bp, bp + 16);
            acc[j] = wmma_f16(A, B, acc[j]);
        }
    }
#pragma unroll
    for (int j = 0; j < 4; j++) {
        int n = n0 + j * 16 + nl;
        float bn = bo[n];
#pragma unroll
        for (int r = 0; r < 8; r++) {
            size_t m = (size_t)mt * 16 + r + 8 * h;
            out[m * DIMX + n] = acc[j][r] + bn + x[m * DIMX + n];
        }
    }
}

extern "C" void kernel_launch(void* const* d_in, const int* in_sizes, int n_in,
                              void* d_out, int out_size, void* d_ws, size_t ws_size,
                              hipStream_t stream) {
    const float* x        = (const float*)d_in[0];
    const float* qk_s     = (const float*)d_in[1];
    const float* hidden_s = (const float*)d_in[2];
    const float* out_s    = (const float*)d_in[3];
    const float* ln_g     = (const float*)d_in[4];
    const float* ln_b     = (const float*)d_in[5];
    const float* Wh       = (const float*)d_in[6];
    const float* bh       = (const float*)d_in[7];
    const float* Wqk      = (const float*)d_in[8];
    const float* bq       = (const float*)d_in[9];
    const float* osg      = (const float*)d_in[10];
    const float* osb      = (const float*)d_in[11];
    const float* rbt      = (const float*)d_in[12];
    const float* Wo       = (const float*)d_in[13];
    const float* bo       = (const float*)d_in[14];

    char* ws = (char*)d_ws;
    size_t o = 0;
    _Float16* normed  = (_Float16*)(ws + o); o += (size_t)BATCH * SEQ * DIMX * 2;   // 16 MB
    _Float16* WhT     = (_Float16*)(ws + o); o += (size_t)2048 * 512 * 2;           //  2 MB
    _Float16* WqkT    = (_Float16*)(ws + o); o += (size_t)128 * 512 * 2;            // 128 KB
    _Float16* WoT     = (_Float16*)(ws + o); o += (size_t)512 * 1024 * 2;           //  1 MB
    float*    biasTab = (float*)   (ws + o); o += (size_t)16384;                    // 16 KB
    float*    qk32    = (float*)   (ws + o); o += (size_t)BATCH * SEQ * QKD * 4;    //  8 MB
    _Float16* qh      = (_Float16*)(ws + o); o += (size_t)BATCH * SEQ * QKD * 2;    //  4 MB
    _Float16* kh      = (_Float16*)(ws + o); o += (size_t)BATCH * SEQ * QKD * 2;    //  4 MB
    _Float16* vT      = (_Float16*)(ws + o); o += (size_t)BATCH * HID * SEQ * 2;    // 32 MB
    _Float16* gateh   = (_Float16*)(ws + o); o += (size_t)BATCH * SEQ * HID * 2;    // 32 MB
    _Float16* outp    = (_Float16*)(ws + o); o += (size_t)BATCH * SEQ * HID * 2;    // 32 MB

    // weight prep + bias table
    transpose_f16_kernel<<<(512 * 2048 + 255) / 256, 256, 0, stream>>>(Wh, WhT, 512, 2048);
    transpose_f16_kernel<<<(512 * 128 + 255) / 256, 256, 0, stream>>>(Wqk, WqkT, 512, 128);
    transpose_f16_kernel<<<(1024 * 512 + 255) / 256, 256, 0, stream>>>(Wo, WoT, 1024, 512);
    bias_table_kernel<<<(2 * SEQ - 1 + 255) / 256, 256, 0, stream>>>(rbt, biasTab);

    // LN + shift + scale
    ln_shift_kernel<<<BATCH * SEQ, 256, 0, stream>>>(x, ln_g, ln_b, qk_s, hidden_s, normed);

    // hidden GEMM (v transposed + gate)
    gemm_hidden_kernel<<<(1024 * 32) / 8, 256, 0, stream>>>(normed, WhT, bh, vT, gateh);

    // qk GEMM + rotary
    gemm_qk_kernel<<<(1024 * 2) / 8, 256, 0, stream>>>(normed, WqkT, bq, qk32);
    rotary_kernel<<<BATCH * SEQ, 128, 0, stream>>>(qk32, osg, osb, qh, kh);

    // fused attention
    attn_kernel<<<BATCH * (SEQ / 16), 256, 0, stream>>>(qh, kh, vT, gateh, biasTab, out_s, outp);

    // output projection + residual
    gemm_out_kernel<<<(1024 * 8) / 8, 256, 0, stream>>>(outp, WoT, bo, x, (float*)d_out);
}
